// SA_6519760355387
// MI455X (gfx1250) — matrix operand order repaired and checked by reading, not compiled, not run on hardware
//
#include <hip/hip_runtime.h>

#define DEV static __device__ __forceinline__

#if defined(__has_builtin)
#if __has_builtin(__builtin_amdgcn_sched_barrier)
#define SCHED_FENCE() __builtin_amdgcn_sched_barrier(0)
#endif
#endif
#ifndef SCHED_FENCE
#define SCHED_FENCE()
#endif

typedef __attribute__((ext_vector_type(16))) __bf16 v16bf;
typedef __attribute__((ext_vector_type(8)))  float  v8f;

constexpr int Dm = 1024, Hh = 16, HDd = 64, Bb = 2, Ss = 2048, Mm = Bb * Ss;

DEV unsigned short f2bf(float f) {
  unsigned int u = __float_as_uint(f);
  unsigned int r = (u + 0x7FFFu + ((u >> 16) & 1u)) >> 16;  // RNE
  return (unsigned short)r;
}

DEV v8f wmma_bf16(v16bf a, v16bf b, v8f c) {
  return __builtin_amdgcn_wmma_f32_16x16x32_bf16(
      /*neg_a=*/false, a, /*neg_b=*/false, b,
      /*c_mod=*/(short)0, c, /*reuse_a=*/false, /*reuse_b=*/false);
}

// A-operand fragment (16x32 bf16): lane reads 16 contiguous K elems at p
// (caller pre-applies kb = lane<16 ? 0 : 16).
DEV v16bf ld_frag_split(const unsigned short* p) {
  union { uint4 q[2]; v16bf v; } u;
  u.q[0] = *(const uint4*)(p);
  u.q[1] = *(const uint4*)(p + 8);
  return u.v;
}
// B-operand fragment (32x16 bf16): lane reads 16 contiguous K elems at p
// (caller pre-applies kb = lane<16 ? 0 : 16).
DEV v16bf ld_frag_contig(const unsigned short* p) {
  union { uint4 q[2]; v16bf v; } u;
  u.q[0] = *(const uint4*)(p);
  u.q[1] = *(const uint4*)(p + 8);
  return u.v;
}

// ---------------------------------------------------------------- converts
// 4 floats -> 4 bf16 per thread (b128 in, b64 out)
__global__ __launch_bounds__(256) void cvt_bf16_kernel(
    const float4* __restrict__ src, uint2* __restrict__ dst, int n4) {
  int i = blockIdx.x * 256 + threadIdx.x;
  if (i < n4) {
    float4 f = src[i];
    uint2 o;
    o.x = (unsigned int)f2bf(f.x) | ((unsigned int)f2bf(f.y) << 16);
    o.y = (unsigned int)f2bf(f.z) | ((unsigned int)f2bf(f.w) << 16);
    dst[i] = o;
  }
}

// ---------------------------------------------------------------- GEMM
// C[M,N] = A[M,K](bf16) * W[K,N](bf16) + bias ; k-step 64, 8 WMMAs / barrier pair
// mode 0: store bf16 scattered to [B,H,S,hd]  (QKV projections)
// mode 1: store f32 row-major [M,N]           (output projection)
__global__ __launch_bounds__(256) void gemm_bf16_kernel(
    const unsigned short* __restrict__ A, const unsigned short* __restrict__ Wb,
    const float* __restrict__ bias, unsigned short* __restrict__ out_bf,
    float* __restrict__ out_f, int mode, int M, int N, int K) {
  // stride 72 halfwords = 36 dwords: 36*n mod 64 = 4*(9n mod 16) -> conflict-free
  __shared__ __align__(16) unsigned short As[128 * 72];
  __shared__ __align__(16) unsigned short Wt[64 * 72];

  const int t = threadIdx.x, lane = t & 31, w = t >> 5;
  const int m0 = blockIdx.y * 128, n0 = blockIdx.x * 64;
  const int ln = lane & 15;
  const int h0 = (lane < 16) ? 0 : 8;
  const int kb = (lane < 16) ? 0 : 16;
  const int grp8 = (lane < 16) ? 0 : 8;

  v8f acc[4];
  for (int i = 0; i < 4; i++)
    for (int j = 0; j < 8; j++) acc[i][j] = 0.f;

  const int arow_ld = t >> 1, ahalf = t & 1;       // A: 128 rows x 64 K
  const int wrow_ld = t >> 2, wn16 = (t & 3) * 16; // W: 64 K x 64 N

  for (int k0 = 0; k0 < K; k0 += 64) {
    __syncthreads();
    {  // stage A tile [128][64] -> As (row-major, stride 72)
      const uint4* src = (const uint4*)(A + (size_t)(m0 + arow_ld) * K + k0 + ahalf * 32);
      uint4 a0 = src[0], a1 = src[1], a2 = src[2], a3 = src[3];
      uint4* dst = (uint4*)&As[arow_ld * 72 + ahalf * 32];
      dst[0] = a0; dst[1] = a1; dst[2] = a2; dst[3] = a3;
    }
    {  // stage W tile transposed: Wt[n][k], 64 x 64
      const uint4* src = (const uint4*)(Wb + (size_t)(k0 + wrow_ld) * N + n0 + wn16);
      uint4 w0 = src[0], w1 = src[1];
      const unsigned short* pw0 = (const unsigned short*)&w0;
      const unsigned short* pw1 = (const unsigned short*)&w1;
#pragma unroll
      for (int i = 0; i < 8; i++) Wt[(wn16 + i) * 72 + wrow_ld] = pw0[i];
#pragma unroll
      for (int i = 0; i < 8; i++) Wt[(wn16 + 8 + i) * 72 + wrow_ld] = pw1[i];
    }
    if (k0 + 64 < K)  // global_prefetch_b8 for next A tile
      __builtin_prefetch(A + (size_t)(m0 + arow_ld) * K + k0 + 64, 0, 1);
    __syncthreads();

    // preload ALL fragments (one ds clause, one wait) ...
    const unsigned short* arow = &As[(w * 16 + ln) * 72];
    v16bf a0 = ld_frag_split(arow + kb);         // K 0..31
    v16bf a1 = ld_frag_split(arow + 32 + kb);    // K 32..63
    v16bf bfr[8];
#pragma unroll
    for (int nf = 0; nf < 4; nf++) {
      const unsigned short* brow = &Wt[(nf * 16 + ln) * 72];
      bfr[2 * nf]     = ld_frag_contig(brow + kb);
      bfr[2 * nf + 1] = ld_frag_contig(brow + 32 + kb);
    }
    SCHED_FENCE();
    // ... then stream 8 dependency-free WMMAs back-to-back on the XDL pipe
#pragma unroll
    for (int nf = 0; nf < 4; nf++) {
      acc[nf] = wmma_bf16(a0, bfr[2 * nf], acc[nf]);
      acc[nf] = wmma_bf16(a1, bfr[2 * nf + 1], acc[nf]);
    }
    SCHED_FENCE();
  }

#pragma unroll
  for (int nf = 0; nf < 4; nf++) {
    int ncol = n0 + nf * 16 + ln;
    float bv = bias[ncol];
#pragma unroll
    for (int r = 0; r < 8; r++) {
      int mrow = m0 + w * 16 + r + grp8;   // C layout: VGPR r -> M = r (+8 for hi lanes)
      float vv = acc[nf][r] + bv;
      if (mode == 0) {
        int b = mrow >> 11, s = mrow & (Ss - 1);
        int h = ncol >> 6, hd = ncol & (HDd - 1);
        out_bf[(((size_t)(b * Hh + h)) * Ss + s) * HDd + hd] = f2bf(vv);
      } else {
        out_f[(size_t)mrow * N + ncol] = vv;
      }
    }
  }
}

// ---------------------------------------------------------------- attention
// Q,K,V: bf16 [B*H, S, 64]; Ob: bf16 [B, S, D] (head-major cols) for final GEMM.
__global__ __launch_bounds__(128) void attn_kernel(
    const unsigned short* __restrict__ Q, const unsigned short* __restrict__ Kv,
    const unsigned short* __restrict__ Vv, const int* __restrict__ amask,
    unsigned short* __restrict__ Ob) {
  __shared__ __align__(16) unsigned short Qs[64 * 72];     // Q block [row][d]
  __shared__ __align__(16) unsigned short Ks[32 * 72];     // K tile  [j][d]
  __shared__ __align__(16) unsigned short Vt[64 * 40];     // V tile transposed [d][j]
  __shared__ __align__(16) unsigned short Ps[4 * 16 * 40]; // per-wave P patch

  const int t = threadIdx.x, lane = t & 31, w = t >> 5;
  const int ln = lane & 15;
  const int h0 = (lane < 16) ? 0 : 8;
  const int kb = (lane < 16) ? 0 : 16;
  const int grp8 = (lane < 16) ? 0 : 8;

  const int q0 = blockIdx.x * 64;
  const int bh = blockIdx.y;
  const int bidx = bh >> 4;    // H == 16
  const float scale = 0.125f;  // 1/sqrt(64)
  const float NEG_INF = -__builtin_inff();

  {  // stage 64x64 Q block once
    int row = t >> 1, half = t & 1;
    const uint4* src = (const uint4*)(Q + ((size_t)bh * Ss + q0 + row) * HDd + half * 32);
    uint4 q4[4] = {src[0], src[1], src[2], src[3]};
    uint4* dst = (uint4*)&Qs[row * 72 + half * 32];
    dst[0] = q4[0]; dst[1] = q4[1]; dst[2] = q4[2]; dst[3] = q4[3];
  }
  __syncthreads();

  const unsigned short* qrow = &Qs[(w * 16 + ln) * 72];
  v16bf aq0 = ld_frag_split(qrow + kb);        // d = 0..31
  v16bf aq1 = ld_frag_split(qrow + 32 + kb);   // d = 32..63

  float m_st[8], l_st[8], alph[8];
  v8f oacc[4];
  for (int r = 0; r < 8; r++) { m_st[r] = NEG_INF; l_st[r] = 0.f; }
  for (int i = 0; i < 4; i++)
    for (int j = 0; j < 8; j++) oacc[i][j] = 0.f;

  const int stage_j = t >> 2, stage_h = t & 3;

  for (int kt = 0; kt < q0 + 64; kt += 32) {  // causal: only tiles touching diag
    __syncthreads();
    {  // stage K tile [32][64]
      const uint4* src = (const uint4*)(Kv + ((size_t)bh * Ss + kt + stage_j) * HDd + stage_h * 16);
      uint4 k0 = src[0], k1 = src[1];
      uint4* dst = (uint4*)&Ks[stage_j * 72 + stage_h * 16];
      dst[0] = k0; dst[1] = k1;
    }
    {  // stage V tile transposed Vt[d][j]
      const uint4* src = (const uint4*)(Vv + ((size_t)bh * Ss + kt + stage_j) * HDd + stage_h * 16);
      uint4 v0 = src[0], v1 = src[1];
      const unsigned short* pv0 = (const unsigned short*)&v0;
      const unsigned short* pv1 = (const unsigned short*)&v1;
#pragma unroll
      for (int i = 0; i < 8; i++) Vt[(stage_h * 16 + i) * 40 + stage_j] = pv0[i];
#pragma unroll
      for (int i = 0; i < 8; i++) Vt[(stage_h * 16 + 8 + i) * 40 + stage_j] = pv1[i];
    }
    __syncthreads();

    // S = Q*K^T : preload 4 B-fragments, then 4 WMMAs back-to-back
    v16bf bk[4];
#pragma unroll
    for (int hh = 0; hh < 2; hh++) {
      const unsigned short* krow = &Ks[(hh * 16 + ln) * 72];
      bk[2 * hh]     = ld_frag_contig(krow + kb);        // d 0..31
      bk[2 * hh + 1] = ld_frag_contig(krow + 32 + kb);   // d 32..63
    }
    SCHED_FENCE();
    v8f sc[2];
#pragma unroll
    for (int hh = 0; hh < 2; hh++) {
      v8f z; for (int j = 0; j < 8; j++) z[j] = 0.f;
      z = wmma_bf16(aq0, bk[2 * hh], z);
      z = wmma_bf16(aq1, bk[2 * hh + 1], z);
      sc[hh] = z;
    }
    SCHED_FENCE();

    int am0 = amask[(size_t)bidx * Ss + kt + ln];
    int am1 = amask[(size_t)bidx * Ss + kt + 16 + ln];
    int kj0 = kt + ln, kj1 = kt + 16 + ln;

#pragma unroll
    for (int r = 0; r < 8; r++) {
      int qi = q0 + w * 16 + r + grp8;
      float s0v = (kj0 <= qi && am0 != 0) ? sc[0][r] * scale : NEG_INF;
      float s1v = (kj1 <= qi && am1 != 0) ? sc[1][r] * scale : NEG_INF;
      float rm = fmaxf(s0v, s1v);
#pragma unroll
      for (int off = 1; off < 16; off <<= 1) rm = fmaxf(rm, __shfl_xor(rm, off, 16));
      float mnew = fmaxf(m_st[r], rm);          // kt=0 always has valid keys -> finite
      float al = __expf(m_st[r] - mnew);
      float p0 = __expf(s0v - mnew);
      float p1 = __expf(s1v - mnew);
      float rs = p0 + p1;
#pragma unroll
      for (int off = 1; off < 16; off <<= 1) rs += __shfl_xor(rs, off, 16);
      l_st[r] = l_st[r] * al + rs;
      m_st[r] = mnew;
      alph[r] = al;
      // scatter P (C layout) into per-wave LDS patch
      Ps[(w * 16 + r + grp8) * 40 + ln] = f2bf(p0);
      Ps[(w * 16 + r + grp8) * 40 + 16 + ln] = f2bf(p1);
    }
#pragma unroll
    for (int nf = 0; nf < 4; nf++)
#pragma unroll
      for (int r = 0; r < 8; r++) oacc[nf][r] *= alph[r];

    __syncthreads();  // P patch visible (also keeps waves in lockstep for staging)

    // O += P * V : preload A + 4 B-fragments, then 4 WMMAs back-to-back
    v16bf ap = ld_frag_split(&Ps[(w * 16 + ln) * 40 + kb]);
    v16bf bvf[4];
#pragma unroll
    for (int nf = 0; nf < 4; nf++)
      bvf[nf] = ld_frag_contig(&Vt[(nf * 16 + ln) * 40 + kb]);
    SCHED_FENCE();
#pragma unroll
    for (int nf = 0; nf < 4; nf++)
      oacc[nf] = wmma_bf16(ap, bvf[nf], oacc[nf]);
    SCHED_FENCE();
  }

  const int hidx = bh & (Hh - 1);
#pragma unroll
  for (int nf = 0; nf < 4; nf++) {
#pragma unroll
    for (int r = 0; r < 8; r++) {
      int qi = q0 + w * 16 + r + grp8;
      int d = nf * 16 + ln;
      float ov = oacc[nf][r] / l_st[r];
      Ob[((size_t)(bidx * Ss + qi)) * Dm + hidx * HDd + d] = f2bf(ov);
    }
  }
}

// ---------------------------------------------------------------- launch
extern "C" void kernel_launch(void* const* d_in, const int* in_sizes, int n_in,
                              void* d_out, int out_size, void* d_ws, size_t ws_size,
                              hipStream_t stream) {
  (void)in_sizes; (void)n_in; (void)out_size; (void)ws_size;
  const float* x  = (const float*)d_in[0];
  const int*  am  = (const int*)d_in[1];
  const float* Wq = (const float*)d_in[2]; const float* bq = (const float*)d_in[3];
  const float* Wk = (const float*)d_in[4]; const float* bk = (const float*)d_in[5];
  const float* Wv = (const float*)d_in[6]; const float* bv = (const float*)d_in[7];
  const float* Wo = (const float*)d_in[8]; const float* bo = (const float*)d_in[9];

  char* ws = (char*)d_ws;
  size_t off = 0;
  auto take = [&](size_t elems) -> unsigned short* {
    unsigned short* p = (unsigned short*)(ws + off);
    off += ((elems * 2 + 255) & ~(size_t)255);
    return p;
  };
  unsigned short* xb  = take((size_t)Mm * Dm);
  unsigned short* wqb = take((size_t)Dm * Dm);
  unsigned short* wkb = take((size_t)Dm * Dm);
  unsigned short* wvb = take((size_t)Dm * Dm);
  unsigned short* wob = take((size_t)Dm * Dm);
  unsigned short* qb  = take((size_t)Mm * Dm);
  unsigned short* kbf = take((size_t)Mm * Dm);
  unsigned short* vbf = take((size_t)Mm * Dm);
  unsigned short* ob  = take((size_t)Mm * Dm);

  const int nx4 = Mm * Dm / 4;  // 1048576
  const int nw4 = Dm * Dm / 4;  // 262144
  cvt_bf16_kernel<<<nx4 / 256, 256, 0, stream>>>((const float4*)x,  (uint2*)xb,  nx4);
  cvt_bf16_kernel<<<nw4 / 256, 256, 0, stream>>>((const float4*)Wq, (uint2*)wqb, nw4);
  cvt_bf16_kernel<<<nw4 / 256, 256, 0, stream>>>((const float4*)Wk, (uint2*)wkb, nw4);
  cvt_bf16_kernel<<<nw4 / 256, 256, 0, stream>>>((const float4*)Wv, (uint2*)wvb, nw4);
  cvt_bf16_kernel<<<nw4 / 256, 256, 0, stream>>>((const float4*)Wo, (uint2*)wob, nw4);

  dim3 gg(Dm / 64, Mm / 128);  // (16, 32)
  gemm_bf16_kernel<<<gg, 256, 0, stream>>>(xb, wqb, bq, qb,  nullptr, 0, Mm, Dm, Dm);
  gemm_bf16_kernel<<<gg, 256, 0, stream>>>(xb, wkb, bk, kbf, nullptr, 0, Mm, Dm, Dm);
  gemm_bf16_kernel<<<gg, 256, 0, stream>>>(xb, wvb, bv, vbf, nullptr, 0, Mm, Dm, Dm);

  attn_kernel<<<dim3(Ss / 64, Bb * Hh), 128, 0, stream>>>(qb, kbf, vbf, am, ob);

  gemm_bf16_kernel<<<gg, 256, 0, stream>>>(ob, wob, bo, nullptr, (float*)d_out, 1, Mm, Dm, Dm);
}
